// RNNModel_51926154609339
// MI455X (gfx1250) — matrix operand order repaired and checked by reading, not compiled, not run on hardware
//
#include <hip/hip_runtime.h>
#include <hip/hip_bf16.h>
#include <math.h>

// ---------------------------------------------------------------------------
// RNN: out = softmax( tanh_scan( X@Wx + b , Wh ) @ Wd + bd )
// B=32, T=512, V=1024, H=256.  GEMMs via v_wmma_f32_16x16x32_bf16,
// Wh^T -> LDS via the Tensor Data Mover, tanh via V_TANH_F32.
// ---------------------------------------------------------------------------

typedef __attribute__((ext_vector_type(16))) __bf16 bf16x16;
typedef __attribute__((ext_vector_type(8)))  float  v8f;
typedef __attribute__((ext_vector_type(4)))  unsigned int u32x4;
typedef __attribute__((ext_vector_type(8)))  int i32x8;
typedef __attribute__((ext_vector_type(4)))  int i32x4;

#define Bsz 32
#define Tsz 512
#define Vsz 1024
#define Hsz 256

__device__ __forceinline__ v8f wmma_bf16(bf16x16 a, bf16x16 b, v8f c) {
    return __builtin_amdgcn_wmma_f32_16x16x32_bf16(
        /*neg_a=*/false, a, /*neg_b=*/false, b,
        /*c_mod=*/(short)0, c, /*reuse_a=*/false, /*reuse_b=*/false);
}

__device__ __forceinline__ float fast_tanh(float x) {
#if __has_builtin(__builtin_amdgcn_tanhf)
    return __builtin_amdgcn_tanhf(x);      // V_TANH_F32 (CDNA5 trans op)
#elif __has_builtin(__builtin_amdgcn_tanh_f32)
    return __builtin_amdgcn_tanh_f32(x);
#else
    return tanhf(x);
#endif
}

// A operand (16x32 bf16, M x K) loaded from row-major f32, row stride ld.
// ISA layout: lane L (grp g=L/16, row=L%16); VGPR v<4: K=8g+2v+{0,1};
//             v>=4: K=16+8g+2(v-4)+{0,1}.
__device__ __forceinline__ bf16x16 load_A_f32(const float* Ab, int ld, int lane) {
    int g = lane >> 4, row = lane & 15;
    const float* rp = Ab + row * ld;
    bf16x16 a;
#pragma unroll
    for (int v = 0; v < 8; ++v) {
        int k0 = (v < 4) ? (8 * g + 2 * v) : (16 + 8 * g + 2 * (v - 4));
        a[2 * v]     = (__bf16)rp[k0];
        a[2 * v + 1] = (__bf16)rp[k0 + 1];
    }
    return a;
}

__device__ __forceinline__ bf16x16 load_A_bf16(const __bf16* Ab, int ld, int lane) {
    int g = lane >> 4, row = lane & 15;
    const __bf16* rp = Ab + row * ld;
    bf16x16 a;
#pragma unroll
    for (int v = 0; v < 8; ++v) {
        int k0 = (v < 4) ? (8 * g + 2 * v) : (16 + 8 * g + 2 * (v - 4));
        a[2 * v]     = rp[k0];
        a[2 * v + 1] = rp[k0 + 1];
    }
    return a;
}

// B operand (32x16 bf16, K x N) from K-contiguous transposed weight Wt[n][k]
// (row stride ld).  Lane L holds column n=L%16, g=L/16; j<8: K=8g+j; j>=8: K=16+8g+(j-8).
__device__ __forceinline__ bf16x16 load_B_T(const __bf16* Wt, int ld, int lane) {
    int g = lane >> 4, n = lane & 15;
    const __bf16* p = Wt + n * ld + 8 * g;
    bf16x16 b;
#pragma unroll
    for (int j = 0; j < 16; ++j) b[j] = p[(j < 8) ? j : (8 + j)];
    return b;
}

// ---------------------------------------------------------------------------
// Kernel 0: transpose + convert weights to bf16 [N][K] (WMMA-B friendly)
// ---------------------------------------------------------------------------
__global__ void __launch_bounds__(256)
pack_weights(const float* __restrict__ Wx, const float* __restrict__ Wh,
             const float* __restrict__ Wd,
             __bf16* __restrict__ WxT, __bf16* __restrict__ WhT,
             __bf16* __restrict__ WdT) {
    int stride = gridDim.x * 256;
    int idx = blockIdx.x * 256 + threadIdx.x;
    for (int i = idx; i < Vsz * Hsz; i += stride) {       // WxT[n][k]=Wx[k][n]
        int n = i >> 10, k = i & 1023;
        WxT[i] = (__bf16)Wx[k * Hsz + n];
    }
    for (int i = idx; i < Hsz * Hsz; i += stride) {       // WhT[n][k]=Wh[k][n]
        int n = i >> 8, k = i & 255;
        WhT[i] = (__bf16)Wh[k * Hsz + n];
    }
    for (int i = idx; i < Hsz * Vsz; i += stride) {       // WdT[n][k]=Wd[k][n]
        int n = i >> 8, k = i & 255;
        WdT[i] = (__bf16)Wd[k * Vsz + n];
    }
}

// ---------------------------------------------------------------------------
// Kernel 1: xproj[t][b][h] = (X @ Wx)[b*T+t][h] + bias[h]
// 16384 (M) x 256 (N), K=1024.  Wave = 1 M-tile x 4 N-tiles (A reused 4x).
// ---------------------------------------------------------------------------
__global__ void __launch_bounds__(256)
xproj_kernel(const float* __restrict__ X, const __bf16* __restrict__ WxT,
             const float* __restrict__ bias, float* __restrict__ xproj) {
    int gw   = blockIdx.x * 8 + (threadIdx.x >> 5); // 0..4095
    int mt   = gw >> 2;                             // 0..1023
    int ntb  = (gw & 3) * 4;                        // 0,4,8,12
    int lane = threadIdx.x & 31;

    const float* Ab = X + (size_t)(mt * 16) * Vsz;

    v8f acc[4] = {};
    for (int k = 0; k < Vsz; k += 32) {
        bf16x16 a = load_A_f32(Ab + k, Vsz, lane);
#pragma unroll
        for (int q = 0; q < 4; ++q) {
            bf16x16 b = load_B_T(WxT + (size_t)(ntb + q) * 16 * Vsz + k, Vsz, lane);
            acc[q] = wmma_bf16(a, b, acc[q]);
        }
    }

    int g = lane >> 4, col = lane & 15;
#pragma unroll
    for (int q = 0; q < 4; ++q) {
        int n = (ntb + q) * 16 + col;
        float bn = bias[n];
#pragma unroll
        for (int i = 0; i < 8; ++i) {
            int m  = mt * 16 + i + 8 * g;   // m = b*T + t
            int bb = m >> 9;                // / T
            int t  = m & (Tsz - 1);
            xproj[((size_t)t * Bsz + bb) * Hsz + n] = acc[q][i] + bn;
        }
    }
}

// ---------------------------------------------------------------------------
// Kernel 2: sequential scan, ONE workgroup.  Wh^T (128KB bf16) + h (16KB bf16)
// live in LDS (CDNA5: 320KB/WGP).  Wh^T filled by the Tensor Data Mover.
// ---------------------------------------------------------------------------
__global__ void __launch_bounds__(256)
rnn_scan_kernel(const float* __restrict__ xproj, const __bf16* __restrict__ WhT_g,
                float* __restrict__ state_out, __bf16* __restrict__ Ybf) {
    __shared__ __bf16 WhT[Hsz * Hsz];  // 128 KB
    __shared__ __bf16 hS[Bsz * Hsz];   //  16 KB

    int tid = threadIdx.x;
    int w = tid >> 5, lane = tid & 31;

#if __has_builtin(__builtin_amdgcn_tensor_load_to_lds) && defined(__gfx1250__)
    if (w == 0) {
        // Tensor DMA descriptor (D#): 1-D tile of 32768 DWORDs (=128KB Wh^T).
        unsigned long long ga = (unsigned long long)(uintptr_t)WhT_g;
        unsigned lds_off = (unsigned)(uintptr_t)(void*)&WhT[0]; // LDS aperture: low 32b
        u32x4 g0;
        g0[0] = 1u;                                   // count=1 (valid), user mode
        g0[1] = lds_off;                              // lds_addr (bytes)
        g0[2] = (unsigned)(ga & 0xFFFFFFFFu);         // global_addr[31:0]
        g0[3] = (unsigned)((ga >> 32) & 0x01FFFFFFu)  // global_addr[56:32]
              | (2u << 30);                           // type=2 ("image")
        i32x8 g1;
        g1[0] = 2 << 16;          // workgroup_mask=0, data_size=2 (4 bytes)
        g1[1] = (int)0x80000000;  // tensor_dim0[15:0]=0x8000 (32768 DWORDs)
        g1[2] = 1 << 16;          // tensor_dim0[31:16]=0, tensor_dim1=1
        g1[3] = (int)0x80000000;  // tile_dim0=0x8000 in bits[31:16]
        g1[4] = 1;                // tile_dim1=1, tile_dim2=0
        g1[5] = 32768;            // tensor_dim0_stride (DWORDs)
        g1[6] = 0;                // stride hi / tensor_dim1_stride lo
        g1[7] = 0;
        i32x4 z4 = {0, 0, 0, 0};              // groups 2/3 unused (<=2D tensor)
        i32x8 z8 = {0, 0, 0, 0, 0, 0, 0, 0};  // 6-arg toolchain: extra group
        __builtin_amdgcn_tensor_load_to_lds(g0, g1, z4, z4, z8, 0);
#if __has_builtin(__builtin_amdgcn_s_wait_tensorcnt)
        __builtin_amdgcn_s_wait_tensorcnt(0);
#endif
    }
#else
    {   // fallback: cooperative vector copy
        const unsigned long long* src = (const unsigned long long*)WhT_g;
        unsigned long long* dst = (unsigned long long*)WhT;
        for (int i = tid; i < (Hsz * Hsz) / 4; i += 256) dst[i] = src[i];
    }
#endif
    {   // zero running state h
        unsigned long long* hz = (unsigned long long*)hS;
        for (int i = tid; i < (Bsz * Hsz) / 4; i += 256) hz[i] = 0ull;
    }
    __syncthreads();

    int mt     = w >> 2;          // 0..1  (M tile)
    int ntbase = (w & 3) * 4;     // 4 N tiles per wave
    int g   = lane >> 4;
    int col = lane & 15;

    for (int t = 0; t < Tsz; ++t) {
        const float* xp = xproj + (size_t)t * Bsz * Hsz;
        if (t + 1 < Tsz)  // prefetch next timestep's 32KB slab (128B/thread)
            __builtin_prefetch(xp + Bsz * Hsz + tid * 32, 0, 1);

        v8f acc[4] = {};
        for (int k = 0; k < Hsz; k += 32) {
            bf16x16 a = load_A_bf16(hS + mt * 16 * Hsz + k, Hsz, lane);
#pragma unroll
            for (int q = 0; q < 4; ++q) {
                bf16x16 b = load_B_T(WhT + (ntbase + q) * 16 * Hsz + k, Hsz, lane);
                acc[q] = wmma_bf16(a, b, acc[q]);
            }
        }

        float hv[4][8];
#pragma unroll
        for (int q = 0; q < 4; ++q) {
            int n = (ntbase + q) * 16 + col;
#pragma unroll
            for (int i = 0; i < 8; ++i) {
                int m = mt * 16 + i + 8 * g;            // batch index
                hv[q][i] = fast_tanh(acc[q][i] + xp[m * Hsz + n]);
            }
        }

        __syncthreads();  // everyone done reading hS for this step
#pragma unroll
        for (int q = 0; q < 4; ++q) {
            int n = (ntbase + q) * 16 + col;
#pragma unroll
            for (int i = 0; i < 8; ++i) {
                int m = mt * 16 + i + 8 * g;
                __bf16 hb = (__bf16)hv[q][i];
                hS[m * Hsz + n] = hb;                          // next step's A
                Ybf[((size_t)m * Tsz + t) * Hsz + n] = hb;     // Y row b*T+t
            }
        }
        __syncthreads();

        if (t == Tsz - 1) {   // new_state (f32) -> tail of d_out
#pragma unroll
            for (int q = 0; q < 4; ++q) {
                int n = (ntbase + q) * 16 + col;
#pragma unroll
                for (int i = 0; i < 8; ++i) {
                    int m = mt * 16 + i + 8 * g;
                    state_out[m * Hsz + n] = hv[q][i];
                }
            }
        }
    }
}

// ---------------------------------------------------------------------------
// Kernel 3: out = softmax(Y @ Wd + bd).  Block = 16 rows x full V=1024,
// logits staged in 64KB LDS, fused row softmax.
// ---------------------------------------------------------------------------
__global__ void __launch_bounds__(256)
out_kernel(const __bf16* __restrict__ Ybf, const __bf16* __restrict__ WdT,
           const float* __restrict__ bd, float* __restrict__ out) {
    __shared__ float logits[16 * Vsz];  // 64 KB
    __shared__ float red[16 * 16];

    int blk = blockIdx.x;               // rows blk*16 .. +15
    int tid = threadIdx.x, w = tid >> 5, lane = tid & 31;
    int g = lane >> 4, col = lane & 15;

    const __bf16* Ab = Ybf + (size_t)(blk * 16) * Hsz;

    v8f acc[8] = {};
    for (int k = 0; k < Hsz; k += 32) {
        bf16x16 a = load_A_bf16(Ab + k, Hsz, lane);
#pragma unroll
        for (int q = 0; q < 8; ++q) {
            bf16x16 b = load_B_T(WdT + (w * 8 + q) * 16 * Hsz + k, Hsz, lane);
            acc[q] = wmma_bf16(a, b, acc[q]);
        }
    }
#pragma unroll
    for (int q = 0; q < 8; ++q) {
        int n = (w * 8 + q) * 16 + col;
        float bn = bd[n];
#pragma unroll
        for (int i = 0; i < 8; ++i)
            logits[(i + 8 * g) * Vsz + n] = acc[q][i] + bn;
    }
    __syncthreads();

    // row softmax: 16 threads per row, 64 cols each (stride 16)
    int r = tid >> 4, c0 = tid & 15;
    float mx = -3.0e38f;
    for (int j = 0; j < 64; ++j) mx = fmaxf(mx, logits[r * Vsz + c0 + 16 * j]);
    red[r * 16 + c0] = mx;
    __syncthreads();
    if (c0 == 0) {
        float m2 = red[r * 16];
        for (int j = 1; j < 16; ++j) m2 = fmaxf(m2, red[r * 16 + j]);
        red[r * 16] = m2;
    }
    __syncthreads();
    float rowmax = red[r * 16];
    float s = 0.f;
    for (int j = 0; j < 64; ++j) {
        int c = c0 + 16 * j;
        float e = __expf(logits[r * Vsz + c] - rowmax);
        logits[r * Vsz + c] = e;
        s += e;
    }
    __syncthreads();
    red[r * 16 + c0] = s;
    __syncthreads();
    if (c0 == 0) {
        float t2 = 0.f;
        for (int j = 0; j < 16; ++j) t2 += red[r * 16 + j];
        red[r * 16] = 1.0f / t2;
    }
    __syncthreads();
    float inv = red[r * 16];
    float* orow = out + ((size_t)(blk * 16) + r) * Vsz;
    for (int j = 0; j < 64; ++j) {
        int c = c0 + 16 * j;
        orow[c] = logits[r * Vsz + c] * inv;
    }
}

// ---------------------------------------------------------------------------
extern "C" void kernel_launch(void* const* d_in, const int* in_sizes, int n_in,
                              void* d_out, int out_size, void* d_ws, size_t ws_size,
                              hipStream_t stream) {
    const float* X  = (const float*)d_in[0];   // [B,T,V]
    const float* Wx = (const float*)d_in[1];   // [V,H]
    const float* Wh = (const float*)d_in[2];   // [H,H]
    const float* b  = (const float*)d_in[3];   // [H]
    const float* Wd = (const float*)d_in[4];   // [H,V]
    const float* bd = (const float*)d_in[5];   // [V]
    float* out = (float*)d_out;                // [B*T,V] ++ [B,H]

    char* ws = (char*)d_ws;
    __bf16* WxT   = (__bf16*)(ws);                         // 512 KB
    __bf16* WhT   = (__bf16*)(ws + 524288);                // 128 KB
    __bf16* WdT   = (__bf16*)(ws + 655360);                // 512 KB
    float*  xproj = (float*)(ws + 1179648);                //  16 MB  [T][B][H]
    __bf16* Ybf   = (__bf16*)(ws + 17956864);              //   8 MB  [B*T][H]
    float*  state_out = out + (size_t)Bsz * Tsz * Vsz;

    pack_weights<<<1152, 256, 0, stream>>>(Wx, Wh, Wd, WxT, WhT, WdT);
    xproj_kernel<<<512, 256, 0, stream>>>(X, WxT, b, xproj);
    rnn_scan_kernel<<<1, 256, 0, stream>>>(xproj, WhT, state_out, Ybf);
    out_kernel<<<1024, 256, 0, stream>>>(Ybf, WdT, bd, out);
}
